// BodyTransformer_30880814859145
// MI455X (gfx1250) — matrix operand order repaired and checked by reading, not compiled, not run on hardware
//
#include <hip/hip_runtime.h>

// ---------------------------------------------------------------------------
// BodyTransformer (BoT-Hard) forward for MI455X / gfx1250, bf16 WMMA path.
// B=4096 N=32 D=128 E=256 H=8 F=1024 L=6 DH=32.  ~1.27 TFLOP total.
// GEMMs via v_wmma_f32_16x16x32_bf16 (fp32 accum); LN/softmax fp32.
// Weight tiles staged into LDS by the Tensor Data Mover (tensor_load_to_lds,
// TENSORcnt); A operand streamed from L2 with global_prefetch_b8 lookahead.
// x (residual stream) lives fp32 in d_out (128MB, L2-resident on 192MB L2).
// ---------------------------------------------------------------------------

#define BB   4096
#define NN   32
#define DDIM 128
#define EE   256
#define HH   8
#define FF   1024
#define LLAY 6
#define DH   32
#define MROWS (BB * NN)   // 131072
#define KC   256          // K-chunk staged in LDS per TDM transfer (64KB max)

typedef __attribute__((ext_vector_type(16))) __bf16 v16bf;
typedef __attribute__((ext_vector_type(8)))  float  v8f;
typedef __attribute__((ext_vector_type(4)))  unsigned int v4u;
typedef __attribute__((ext_vector_type(8)))  int v8i;
typedef __attribute__((ext_vector_type(4)))  int v4i;

union BFV { v16bf v; uint4 q[2]; unsigned short s[16]; };

__device__ __forceinline__ unsigned short f2bf(float f) {
  unsigned int u = __float_as_uint(f);
  u += 0x7FFFu + ((u >> 16) & 1u);           // round-to-nearest-even
  return (unsigned short)(u >> 16);
}

// A operand, 16x32 bf16 tile from row-major [rows, ldk] storage.
// ISA layout: lane r=l&15 is row M=r; hi=l>>4 selects K sub-chunks:
// elements 0..7 -> K = k0 + hi*8 + e ; elements 8..15 -> K = k0 + 16 + hi*8 + e.
__device__ __forceinline__ BFV load_a_tile(const unsigned short* base, int ldk,
                                           int rowBase, int k0, int lane) {
  const int r = lane & 15, hi = lane >> 4;
  const unsigned short* p = base + (size_t)(rowBase + r) * ldk + k0 + hi * 8;
  BFV x;
  x.q[0] = *(const uint4*)(p);
  x.q[1] = *(const uint4*)(p + 16);
  return x;
}

// B operand, 32x16 bf16 tile; source stored [O, K] row-major so a lane's
// column (n = l&15) is a contiguous K-run. Same K order as A:
// elements 0..7 -> K = k0 + hi*8 + e ; elements 8..15 -> K = k0 + 16 + hi*8 + e.
__device__ __forceinline__ BFV load_b_tile(const unsigned short* base, int ldk,
                                           int colBase, int k0, int lane) {
  const int c = lane & 15, hi = lane >> 4;
  const unsigned short* p = base + (size_t)(colBase + c) * ldk + k0 + hi * 8;
  BFV x;
  x.q[0] = *(const uint4*)(p);
  x.q[1] = *(const uint4*)(p + 16);
  return x;
}

__device__ __forceinline__ v8f wmma_bf16(BFV a, BFV b, v8f c) {
  return __builtin_amdgcn_wmma_f32_16x16x32_bf16(false, a.v, false, b.v,
                                                 (short)0, c, false, false);
}

// ---------------------------------------------------------------------------
// Tensor Data Mover: DMA a 2-D bf16 tile [rows x cols] (row stride
// strideElems) from global memory into LDS at byte offset lds_off.
// D# built per CDNA5 ISA sect.8: group0 {count=1,
// lds_addr, global_addr(57b), type=2}, group1 {data_size=1(2B), tensor_dim0/1,
// tile_dim0/1, tensor_dim0_stride}, groups 2/3 zero (2-D tile).
// Completion tracked with TENSORcnt.
// ---------------------------------------------------------------------------
__device__ __forceinline__ void tdm_load_2d(const unsigned short* gsrc,
                                            unsigned lds_off, int rows,
                                            int cols, int strideElems) {
  const unsigned long long ga = (unsigned long long)gsrc;
  v4u g0;
  g0[0] = 1u;                                        // count=1, user descriptor
  g0[1] = lds_off;                                   // lds_addr (bytes)
  g0[2] = (unsigned)(ga & 0xffffffffu);              // global_addr[31:0]
  g0[3] = (unsigned)((ga >> 32) & 0x01ffffffu) | (2u << 30);  // [56:32] | type=2
  v8i g1;
  g1[0] = (int)(1u << 16);                           // data_size=1 (2 bytes)
  g1[1] = (int)(((unsigned)cols & 0xffffu) << 16);   // tensor_dim0[15:0] @63:48
  g1[2] = (int)((((unsigned)cols >> 16) & 0xffffu) | // tensor_dim0[31:16]
                (((unsigned)rows & 0xffffu) << 16)); // tensor_dim1[15:0]
  g1[3] = (int)((((unsigned)rows >> 16) & 0xffffu) | // tensor_dim1[31:16]
                (((unsigned)cols & 0xffffu) << 16)); // tile_dim0 @127:112
  g1[4] = (int)((unsigned)rows & 0xffffu);           // tile_dim1; tile_dim2=0
  g1[5] = strideElems;                               // tensor_dim0_stride[31:0]
  g1[6] = 0;                                         // stride[47:32]; dim1_stride
  g1[7] = 0;
  v4i g2 = {};                                       // 2-D: dims 2/3 unused
  v4i g3 = {};
#if defined(__clang_major__) && (__clang_major__ >= 23)
  v8i g4 = {};
  __builtin_amdgcn_tensor_load_to_lds(g0, g1, g2, g3, g4, 0);
#else
  __builtin_amdgcn_tensor_load_to_lds(g0, g1, g2, g3, 0);
#endif
}

// ---------------------------------------------------------------------------
// Generic wave-tiled GEMM: C[M,O] = A[M,K](bf16) * W[O,K]^T(bf16) + bias
// Block = 256 threads = 8 waves (4 M-waves x 2 O-waves) -> 128x128 block tile.
// Weight strip W[blockIdx.y*128 .. +128, kc..kc+KC) staged in LDS via TDM by
// wave 0, consumed by all 8 waves via ds_load_b128. A streamed from global
// with prefetch. Per K-step: 4 global b128 + 8 ds b128 + 8 WMMAs per wave.
// ---------------------------------------------------------------------------
enum { MODE_TOK = 0, MODE_QKV = 1, MODE_OPROJ = 2, MODE_FFN1 = 3, MODE_FFN2 = 4 };

template <int MODE>
__global__ __launch_bounds__(256) void gemm_kernel(
    const unsigned short* __restrict__ A,
    const unsigned short* __restrict__ W,      // [O, K] bf16
    const float* __restrict__ bias,
    const float* __restrict__ bias2,           // tokenizer: pos_emb
    float* __restrict__ xout,                  // fp32 out / residual stream
    unsigned short* __restrict__ bout,         // bf16 out (FFN1)
    unsigned short* __restrict__ qd,
    unsigned short* __restrict__ kd,
    unsigned short* __restrict__ vtd,
    int K, int O) {
  __shared__ __align__(16) unsigned short wsh[128 * KC];   // 64 KB weight stage
  const int lane  = threadIdx.x & 31;
  const int w     = threadIdx.x >> 5;
  const int wm    = w & 3, wo = w >> 2;
  const int mBase = blockIdx.x * 128 + wm * 32;
  const int oBase = blockIdx.y * 128 + wo * 64;
  const unsigned short* Wp = W;
  if (MODE == MODE_TOK) Wp += (size_t)blockIdx.z * O * K;   // per-node weights

  v8f zero = {};
  v8f acc[2][4];
#pragma unroll
  for (int i = 0; i < 2; i++)
#pragma unroll
    for (int j = 0; j < 4; j++) acc[i][j] = zero;

  for (int kc = 0; kc < K; kc += KC) {
    const int cols = (K - kc < KC) ? (K - kc) : KC;         // 128 or 256
    __syncthreads();                       // all readers done with prev chunk
    if (w == 0) {
      // DMA W[blockIdx.y*128 .. +128, kc .. kc+cols) -> LDS offset 0
      tdm_load_2d(Wp + (size_t)(blockIdx.y * 128) * K + kc, 0u, 128, cols, K);
      __builtin_amdgcn_s_wait_tensorcnt(0);
    }
    __syncthreads();                       // staged weights visible to WGP

    for (int k0 = 0; k0 < cols; k0 += 32) {
      // prefetch A one chunk (64 elements = 128B) ahead on the stream
      __builtin_prefetch(A + (size_t)(mBase + (lane & 15)) * K + kc + k0 + 64,
                         0, 3);
      __builtin_prefetch(A + (size_t)(mBase + 16 + (lane & 15)) * K + kc + k0 + 64,
                         0, 3);
      BFV a0 = load_a_tile(A, K, mBase,      kc + k0, lane);
      BFV a1 = load_a_tile(A, K, mBase + 16, kc + k0, lane);
      BFV b0 = load_b_tile(wsh, cols, wo * 64,      k0, lane);
      BFV b1 = load_b_tile(wsh, cols, wo * 64 + 16, k0, lane);
      BFV b2 = load_b_tile(wsh, cols, wo * 64 + 32, k0, lane);
      BFV b3 = load_b_tile(wsh, cols, wo * 64 + 48, k0, lane);
      acc[0][0] = wmma_bf16(a0, b0, acc[0][0]);
      acc[0][1] = wmma_bf16(a0, b1, acc[0][1]);
      acc[0][2] = wmma_bf16(a0, b2, acc[0][2]);
      acc[0][3] = wmma_bf16(a0, b3, acc[0][3]);
      acc[1][0] = wmma_bf16(a1, b0, acc[1][0]);
      acc[1][1] = wmma_bf16(a1, b1, acc[1][1]);
      acc[1][2] = wmma_bf16(a1, b2, acc[1][2]);
      acc[1][3] = wmma_bf16(a1, b3, acc[1][3]);
    }
  }

  // C/D layout: VGPR vi, lane -> m = vi + 8*(lane>>4), n = lane&15.
  const int hi = lane >> 4, c0 = lane & 15;
#pragma unroll
  for (int mt = 0; mt < 2; mt++)
#pragma unroll
    for (int ot = 0; ot < 4; ot++)
#pragma unroll
      for (int vi = 0; vi < 8; vi++) {
        const int m = mBase + mt * 16 + vi + 8 * hi;
        const int o = oBase + ot * 16 + c0;
        float val = acc[mt][ot][vi];
        if (MODE == MODE_TOK) {
          const int n = blockIdx.z;
          val += bias[n * EE + o] + bias2[n * EE + o];
          xout[((size_t)m * NN + n) * EE + o] = val;        // m = batch row b
        } else if (MODE == MODE_QKV) {
          val += bias[o];
          const int b = m >> 5, nn = m & 31;                // m = b*N + node
          const unsigned short bv = f2bf(val);
          if (o < EE) {
            const int h = o >> 5, dh = o & 31;
            qd[(((size_t)b * HH + h) * NN + nn) * DH + dh] = bv;
          } else if (o < 2 * EE) {
            const int e = o - EE, h = e >> 5, dh = e & 31;
            kd[(((size_t)b * HH + h) * NN + nn) * DH + dh] = bv;
          } else {                                          // v stored transposed
            const int e = o - 2 * EE, h = e >> 5, dh = e & 31;
            vtd[(((size_t)b * HH + h) * DH + dh) * NN + nn] = bv;
          }
        } else if (MODE == MODE_OPROJ) {
          val += bias[o];
          xout[(size_t)m * EE + o] += val;                  // residual add
        } else if (MODE == MODE_FFN1) {
          val += bias[o];
          val = fmaxf(val, 0.f);                            // ReLU
          bout[(size_t)m * FF + o] = f2bf(val);
        } else {                                            // MODE_FFN2
          val += bias[o];
          xout[(size_t)m * EE + o] += val;                  // residual add
        }
      }
}

// ---------------------------------------------------------------------------
// LayerNorm: one wave per row of x[M, 256]; fp32 stats, bf16 output.
// ---------------------------------------------------------------------------
__global__ __launch_bounds__(256) void ln_kernel(
    const float* __restrict__ x, const float* __restrict__ g,
    const float* __restrict__ bt, unsigned short* __restrict__ out) {
  const int row  = blockIdx.x * 8 + (threadIdx.x >> 5);
  const int lane = threadIdx.x & 31;
  const float* xr = x + (size_t)row * EE + lane * 8;
  float4 p0 = *(const float4*)(xr);
  float4 p1 = *(const float4*)(xr + 4);
  float v[8] = {p0.x, p0.y, p0.z, p0.w, p1.x, p1.y, p1.z, p1.w};
  float s = 0.f, ss = 0.f;
#pragma unroll
  for (int i = 0; i < 8; i++) { s += v[i]; ss += v[i] * v[i]; }
#pragma unroll
  for (int d = 1; d < 32; d <<= 1) { s += __shfl_xor(s, d); ss += __shfl_xor(ss, d); }
  const float mean = s * (1.f / EE);
  const float var  = ss * (1.f / EE) - mean * mean;
  const float rs   = rsqrtf(var + 1e-5f);
  float4 g0 = *(const float4*)(g + lane * 8);
  float4 g1 = *(const float4*)(g + lane * 8 + 4);
  float4 b0 = *(const float4*)(bt + lane * 8);
  float4 b1 = *(const float4*)(bt + lane * 8 + 4);
  float gg[8] = {g0.x, g0.y, g0.z, g0.w, g1.x, g1.y, g1.z, g1.w};
  float bb[8] = {b0.x, b0.y, b0.z, b0.w, b1.x, b1.y, b1.z, b1.w};
  union { unsigned short s16[8]; uint4 q; } o;
#pragma unroll
  for (int i = 0; i < 8; i++) o.s16[i] = f2bf((v[i] - mean) * rs * gg[i] + bb[i]);
  *(uint4*)(out + (size_t)row * EE + lane * 8) = o.q;
}

// ---------------------------------------------------------------------------
// Attention: one wave per (b,h). q,k: [B,H,N,DH] bf16; v: [B,H,DH,N] bf16.
// Mask staged once per block in LDS. 2x2 WMMA for q*k^T (K=DH=32),
// scale+mask, shfl softmax, LDS transpose of attn into A-layout,
// 2x2 WMMA for attn*v, bf16 output [B*N, E].
// ---------------------------------------------------------------------------
__global__ __launch_bounds__(256) void attn_kernel(
    const unsigned short* __restrict__ q, const unsigned short* __restrict__ k,
    const unsigned short* __restrict__ vt, const float* __restrict__ mask,
    unsigned short* __restrict__ obf) {
  __shared__ __align__(16) unsigned short atile[8][NN * NN];  // 16 KB
  __shared__ float maskSh[NN * NN];                           // 4 KB
  const int w = threadIdx.x >> 5, lane = threadIdx.x & 31;
#pragma unroll
  for (int i = 0; i < 4; i++)                  // 1024 floats, 256 threads
    maskSh[threadIdx.x + 256 * i] = mask[threadIdx.x + 256 * i];
  __syncthreads();

  const int gw = blockIdx.x * 8 + w;
  const int b = gw >> 3, h = gw & 7;
  const unsigned short* qp = q  + ((size_t)b * HH + h) * NN * DH;
  const unsigned short* kp = k  + ((size_t)b * HH + h) * NN * DH;
  const unsigned short* vp = vt + ((size_t)b * HH + h) * DH * NN;
  const int hi = lane >> 4, c0 = lane & 15;

  BFV qa[2], kb[2];
  qa[0] = load_a_tile(qp, DH, 0,  0, lane);
  qa[1] = load_a_tile(qp, DH, 16, 0, lane);
  kb[0] = load_b_tile(kp, DH, 0,  0, lane);   // B col = key index
  kb[1] = load_b_tile(kp, DH, 16, 0, lane);

  v8f zero = {};
  v8f s[2][2];
  s[0][0] = wmma_bf16(qa[0], kb[0], zero);
  s[0][1] = wmma_bf16(qa[0], kb[1], zero);
  s[1][0] = wmma_bf16(qa[1], kb[0], zero);
  s[1][1] = wmma_bf16(qa[1], kb[1], zero);

  const float scale = 0.17677669529663687f;   // 1/sqrt(32)
#pragma unroll
  for (int mt = 0; mt < 2; mt++)
#pragma unroll
    for (int nt = 0; nt < 2; nt++)
#pragma unroll
      for (int vi = 0; vi < 8; vi++) {
        const int m = mt * 16 + vi + 8 * hi;
        const int n = nt * 16 + c0;
        s[mt][nt][vi] = s[mt][nt][vi] * scale + maskSh[m * NN + n];
      }

  // Softmax over 32 keys: each row m lives in lanes of one 16-lane half
  // (two VGPRs); xor-shuffles d=1..8 stay inside the half.
#pragma unroll
  for (int mt = 0; mt < 2; mt++)
#pragma unroll
    for (int vi = 0; vi < 8; vi++) {
      float t0 = s[mt][0][vi], t1 = s[mt][1][vi];
      float mx = fmaxf(t0, t1);
      for (int d = 1; d < 16; d <<= 1) mx = fmaxf(mx, __shfl_xor(mx, d));
      float e0 = __expf(t0 - mx), e1 = __expf(t1 - mx);
      float sm = e0 + e1;
      for (int d = 1; d < 16; d <<= 1) sm += __shfl_xor(sm, d);
      const float inv = 1.f / sm;
      const int m = mt * 16 + vi + 8 * hi;
      atile[w][m * NN + c0]      = f2bf(e0 * inv);
      atile[w][m * NN + 16 + c0] = f2bf(e1 * inv);
    }
  __syncthreads();

  BFV aa[2], vb[2];
  const unsigned short* ap = &atile[w][0];
  aa[0] = load_a_tile(ap, NN, 0,  0, lane);   // A row = query, K = key
  aa[1] = load_a_tile(ap, NN, 16, 0, lane);
  vb[0] = load_b_tile(vp, NN, 0,  0, lane);   // B col = dh, K = key (v^T)
  vb[1] = load_b_tile(vp, NN, 16, 0, lane);

  v8f o00 = wmma_bf16(aa[0], vb[0], zero);
  v8f o01 = wmma_bf16(aa[0], vb[1], zero);
  v8f o10 = wmma_bf16(aa[1], vb[0], zero);
  v8f o11 = wmma_bf16(aa[1], vb[1], zero);
  v8f* oc[2][2] = {{&o00, &o01}, {&o10, &o11}};

#pragma unroll
  for (int mt = 0; mt < 2; mt++)
#pragma unroll
    for (int dt = 0; dt < 2; dt++)
#pragma unroll
      for (int vi = 0; vi < 8; vi++) {
        const int m  = mt * 16 + vi + 8 * hi;   // query node
        const int dh = dt * 16 + c0;
        obf[((size_t)b * NN + m) * EE + h * DH + dh] = f2bf((*oc[mt][dt])[vi]);
      }
}

// ---------------------------------------------------------------------------
// fp32 -> bf16 conversion / transpose helpers (run once per launch; tiny).
// ---------------------------------------------------------------------------
__global__ void cvt_kernel(const float* __restrict__ in,
                           unsigned short* __restrict__ out, int n) {
  int i = blockIdx.x * 256 + threadIdx.x;
  if (i < n) out[i] = f2bf(in[i]);
}
__global__ void cvtT_kernel(const float* __restrict__ in,
                            unsigned short* __restrict__ out, int R, int C) {
  int i = blockIdx.x * 256 + threadIdx.x;           // in [R,C] -> out [C,R]
  if (i < R * C) {
    int r = i / C, c = i % C;
    out[(size_t)c * R + r] = f2bf(in[i]);
  }
}
__global__ void cvt_emb_kernel(const float* __restrict__ in,
                               unsigned short* __restrict__ out) {
  int i = blockIdx.x * 256 + threadIdx.x;           // [N,D,E] -> [N,E,D]
  int n = i >> 15, rem = i & 32767;                 // D*E = 32768
  int d = rem >> 8, e = rem & 255;
  out[((size_t)n * EE + e) * DDIM + d] = f2bf(in[i]);
}

// ---------------------------------------------------------------------------
extern "C" void kernel_launch(void* const* d_in, const int* in_sizes, int n_in,
                              void* d_out, int out_size, void* d_ws, size_t ws_size,
                              hipStream_t stream) {
  (void)in_sizes; (void)n_in; (void)out_size; (void)ws_size;
  const float* obs     = (const float*)d_in[0];
  const float* emb_W   = (const float*)d_in[1];
  const float* emb_b   = (const float*)d_in[2];
  const float* pos_emb = (const float*)d_in[3];
  const float* Wqkv    = (const float*)d_in[4];
  const float* bqkv    = (const float*)d_in[5];
  const float* Wo      = (const float*)d_in[6];
  const float* bo      = (const float*)d_in[7];
  const float* ln1_g   = (const float*)d_in[8];
  const float* ln1_b   = (const float*)d_in[9];
  const float* ln2_g   = (const float*)d_in[10];
  const float* ln2_b   = (const float*)d_in[11];
  const float* W1      = (const float*)d_in[12];
  const float* b1      = (const float*)d_in[13];
  const float* W2      = (const float*)d_in[14];
  const float* b2      = (const float*)d_in[15];
  const float* adjm    = (const float*)d_in[16];
  float* x = (float*)d_out;                         // residual stream, fp32

  char* ws = (char*)d_ws;
  size_t off = 0;
  auto take = [&](size_t bytes) -> char* {
    char* p = ws + off;
    off += (bytes + 255) & ~(size_t)255;
    return p;
  };
  // q|k|vT|o is one contiguous 256MB run; FFN intermediate y aliases it
  // (q/k/vT/o are dead by the time FFN1 runs).
  unsigned short* qb    = (unsigned short*)take((size_t)BB * HH * NN * DH * 2);
  unsigned short* kb    = (unsigned short*)take((size_t)BB * HH * NN * DH * 2);
  unsigned short* vtb   = (unsigned short*)take((size_t)BB * HH * DH * NN * 2);
  unsigned short* ob    = (unsigned short*)take((size_t)MROWS * EE * 2);
  unsigned short* yb    = qb;                       // [MROWS, FF] bf16 alias
  unsigned short* hb    = (unsigned short*)take((size_t)MROWS * EE * 2);
  unsigned short* obsb  = (unsigned short*)take((size_t)BB * DDIM * 2);
  unsigned short* wqkvb = (unsigned short*)take((size_t)3 * EE * EE * 2);
  unsigned short* wob   = (unsigned short*)take((size_t)EE * EE * 2);
  unsigned short* w1tb  = (unsigned short*)take((size_t)FF * EE * 2);
  unsigned short* w2tb  = (unsigned short*)take((size_t)EE * FF * 2);
  unsigned short* embtb = (unsigned short*)take((size_t)NN * EE * DDIM * 2);

  // --- one-time conversions to bf16 (weights transposed to [O, K]) ---
  cvt_kernel<<<(BB * DDIM + 255) / 256, 256, 0, stream>>>(obs, obsb, BB * DDIM);
  cvt_kernel<<<(3 * EE * EE + 255) / 256, 256, 0, stream>>>(Wqkv, wqkvb, 3 * EE * EE);
  cvt_kernel<<<(EE * EE + 255) / 256, 256, 0, stream>>>(Wo, wob, EE * EE);
  cvtT_kernel<<<(EE * FF + 255) / 256, 256, 0, stream>>>(W1, w1tb, EE, FF); // ->[F,E]
  cvtT_kernel<<<(FF * EE + 255) / 256, 256, 0, stream>>>(W2, w2tb, FF, EE); // ->[E,F]
  cvt_emb_kernel<<<(NN * DDIM * EE) / 256, 256, 0, stream>>>(emb_W, embtb);

  // --- tokenizer: x[b,n,:] = obs[b,:] @ embW[n] + emb_b[n] + pos_emb[n] ---
  gemm_kernel<MODE_TOK><<<dim3(BB / 128, EE / 128, NN), 256, 0, stream>>>(
      obsb, embtb, emb_b, pos_emb, x, nullptr, nullptr, nullptr, nullptr,
      DDIM, EE);

  // --- L encoder layers (shared weights) ---
  for (int l = 0; l < LLAY; l++) {
    ln_kernel<<<MROWS / 8, 256, 0, stream>>>(x, ln1_g, ln1_b, hb);
    gemm_kernel<MODE_QKV><<<dim3(MROWS / 128, (3 * EE) / 128), 256, 0, stream>>>(
        hb, wqkvb, bqkv, nullptr, nullptr, nullptr, qb, kb, vtb, EE, 3 * EE);
    attn_kernel<<<(BB * HH) / 8, 256, 0, stream>>>(qb, kb, vtb, adjm, ob);
    gemm_kernel<MODE_OPROJ><<<dim3(MROWS / 128, EE / 128), 256, 0, stream>>>(
        ob, wob, bo, nullptr, x, nullptr, nullptr, nullptr, nullptr, EE, EE);
    ln_kernel<<<MROWS / 8, 256, 0, stream>>>(x, ln2_g, ln2_b, hb);
    gemm_kernel<MODE_FFN1><<<dim3(MROWS / 128, FF / 128), 256, 0, stream>>>(
        hb, w1tb, b1, nullptr, nullptr, yb, nullptr, nullptr, nullptr, EE, FF);
    gemm_kernel<MODE_FFN2><<<dim3(MROWS / 128, EE / 128), 256, 0, stream>>>(
        yb, w2tb, b2, nullptr, x, nullptr, nullptr, nullptr, nullptr, FF, EE);
  }
}
